// EdgeCompute_75428215653108
// MI455X (gfx1250) — matrix operand order repaired and checked by reading, not compiled, not run on hardware
//
#include <hip/hip_runtime.h>
#include <cstdint>

typedef __attribute__((ext_vector_type(16))) _Float16 v16h;
typedef __attribute__((ext_vector_type(8)))  _Float16 v8h;
typedef __attribute__((ext_vector_type(4)))  _Float16 v4h;
typedef __attribute__((ext_vector_type(8)))  float    v8f;
typedef __attribute__((ext_vector_type(4)))  float    v4f;

#define WAVES_PER_BLOCK 8
#define APITCH 136   // halves per A-tile row: 128 + 8 pad (272 B, 16B-aligned, bank-skewed)

__global__ __launch_bounds__(256) void edge_mlp_wmma_kernel(
    const float* __restrict__ x,        // [N_NODES,128] f32
    const long long* __restrict__ idx,  // [2,E] i64
    const float* __restrict__ W1,       // [128,64] f32
    const float* __restrict__ b1,       // [64]
    const float* __restrict__ W2,       // [64]
    const float* __restrict__ b2,       // [1]
    float* __restrict__ out,            // [E]
    int E)
{
    // B fragments of W1, pre-swizzled for v_wmma_f32_16x16x32_f16 B in contiguous k order (same as A):
    // frag(kc,nt,lane) = 16 contiguous halves; element t = W1[kc*32 + (lane/16)*16 + t][nt*16 + (lane%16)]
    __shared__ __align__(16) _Float16 sB[4 * 4 * 32 * 16];              // 16 KB
    __shared__ __align__(16) _Float16 sA[WAVES_PER_BLOCK][16 * APITCH]; // 34 KB
    __shared__ float sB1[64];
    __shared__ float sW2[64];
    __shared__ float sB2v;

    const int tid = threadIdx.x;

    // ---- Phase 0: stage swizzled f16 W1 + small vectors into LDS ----
    for (int f = tid; f < 8192; f += 256) {
        int t    = f & 15;
        int lane = (f >> 4) & 31;
        int nt   = (f >> 9) & 3;
        int kc   = f >> 11;
        int k = kc * 32 + (lane >> 4) * 16 + t;
        int n = nt * 16 + (lane & 15);
        sB[f] = (_Float16)W1[k * 64 + n];
    }
    if (tid < 64) { sB1[tid] = b1[tid]; sW2[tid] = W2[tid]; }
    if (tid == 0) sB2v = b2[0];
    __syncthreads();

    const int lane = tid & 31;
    const int wave = tid >> 5;
    const int row  = lane & 15;   // M row (A frag) / N col (B frag, C/D)
    const int hs   = lane >> 4;   // half-wave select

    // ---- Phase 1: hoist loop-invariant operands into registers ----
    // 16 B fragments (128 VGPRs): W1 stays resident for the whole persistent loop.
    v16h bfrag[4][4];             // [nt][kc]
    #pragma unroll
    for (int nt = 0; nt < 4; ++nt) {
        #pragma unroll
        for (int kc = 0; kc < 4; ++kc) {
            const _Float16* bp = sB + ((kc * 4 + nt) * 32 + lane) * 16;
            v8h lo = *(const v8h*)bp;
            v8h hi = *(const v8h*)(bp + 8);
            v16h bfr;
            #pragma unroll
            for (int i = 0; i < 8; ++i) { bfr[i] = lo[i]; bfr[i + 8] = hi[i]; }
            bfrag[nt][kc] = bfr;
        }
    }
    float biasv[4], w2vv[4];
    #pragma unroll
    for (int nt = 0; nt < 4; ++nt) {
        biasv[nt] = sB1[nt * 16 + row];   // C/D layout: N = lane&15
        w2vv[nt]  = sW2[nt * 16 + row];
    }
    const float b2s = sB2v;

    const int ntiles = (E + 15) >> 4;
    const int gwave  = blockIdx.x * WAVES_PER_BLOCK + wave;
    const int gstep  = gridDim.x * WAVES_PER_BLOCK;

    const int* idx32 = (const int*)idx;       // low dword of each i64 (ids < 2^31)
    const v4f* x4    = (const v4f*)x;         // 32 float4 per 128-f row
    _Float16* aT     = &sA[wave][0];

    for (int tile = gwave; tile < ntiles; tile += gstep) {
        const int ebase = tile << 4;

        // lanes 0..15 each hold one edge's endpoints
        int mySrc = 0, myDst = 0;
        if (lane < 16) {
            int e = ebase + lane;
            if (e < E) {
                mySrc = idx32[2 * e];
                myDst = idx32[2 * (E + e)];
            }
        }

        // ---- build 16x128 f16 |x[s]-x[d]| tile in LDS ----
        #pragma unroll 4
        for (int e = 0; e < 16; ++e) {
            int s = __shfl(mySrc, e, 32);
            int d = __shfl(myDst, e, 32);
            v4f xa = x4[s * 32 + lane];
            v4f xb = x4[d * 32 + lane];
            v4h h;
            #pragma unroll
            for (int i = 0; i < 4; ++i)
                h[i] = (_Float16)__builtin_fabsf(xa[i] - xb[i]);
            *(v4h*)(aT + e * APITCH + lane * 4) = h;
        }
        // wave-local LDS RAW ordering (per-wave tile; no block barrier needed)
        __asm__ volatile("s_wait_dscnt 0" ::: "memory");

        // ---- load 4 A fragments (contiguous k = 16*hs + s order, same as B) ----
        v16h afrag[4];
        #pragma unroll
        for (int kc = 0; kc < 4; ++kc) {
            const _Float16* p = aT + row * APITCH + kc * 32 + hs * 16;
            v8h lo = *(const v8h*)p;         // K {0..7} or {16..23}
            v8h hi = *(const v8h*)(p + 8);   // K {8..15} or {24..31}
            v16h a;
            #pragma unroll
            for (int i = 0; i < 8; ++i) { a[i] = lo[i]; a[i + 8] = hi[i]; }
            afrag[kc] = a;
        }

        float part[8];
        #pragma unroll
        for (int r = 0; r < 8; ++r) part[r] = 0.f;

        // ---- GEMM1: 4 N-tiles x 4 K-chunks, all operands in registers ----
        #pragma unroll
        for (int nt = 0; nt < 4; ++nt) {
            float bias = biasv[nt];
            v8f acc;
            #pragma unroll
            for (int i = 0; i < 8; ++i) acc[i] = bias;

            #pragma unroll
            for (int kc = 0; kc < 4; ++kc) {
                acc = __builtin_amdgcn_wmma_f32_16x16x32_f16(
                    false, afrag[kc], false, bfrag[nt][kc], (short)0, acc, false, false);
            }

            // ReLU + GEMM2 partial: lane covers hidden col n = nt*16 + row
            float w2v = w2vv[nt];
            #pragma unroll
            for (int r = 0; r < 8; ++r) {
                float hv = __builtin_fmaxf(acc[r], 0.f);
                part[r] = __builtin_fmaf(hv, w2v, part[r]);
            }
        }

        // ---- reduce 16 hidden-lanes per half-wave; sigmoid; store ----
        float res = 0.f;
        #pragma unroll
        for (int r = 0; r < 8; ++r) {
            float s = part[r];
            s += __shfl_xor(s, 1, 32);
            s += __shfl_xor(s, 2, 32);
            s += __shfl_xor(s, 4, 32);
            s += __shfl_xor(s, 8, 32);
            if (row == r) res = s;   // edge m = hs*8 + r
        }
        if (row < 8) {
            int e = ebase + hs * 8 + row;
            if (e < E) {
                float z = res + b2s;
                out[e] = 1.0f / (1.0f + __expf(-z));
            }
        }
    }
}

extern "C" void kernel_launch(void* const* d_in, const int* in_sizes, int n_in,
                              void* d_out, int out_size, void* d_ws, size_t ws_size,
                              hipStream_t stream) {
    const float*     x   = (const float*)d_in[0];
    const long long* idx = (const long long*)d_in[1];
    const float*     W1  = (const float*)d_in[2];
    const float*     b1  = (const float*)d_in[3];
    const float*     W2  = (const float*)d_in[4];
    const float*     b2  = (const float*)d_in[5];
    float* out = (float*)d_out;

    const int E = out_size;                       // 640000 edges
    const int ntiles = (E + 15) / 16;
    int blocks = (ntiles + WAVES_PER_BLOCK - 1) / WAVES_PER_BLOCK;
    if (blocks > 640) blocks = 640;               // persistent waves amortize staging + hoisting

    edge_mlp_wmma_kernel<<<blocks, 256, 0, stream>>>(x, idx, W1, b1, W2, b2, out, E);
}
